// JointAggregate_82592221102612
// MI455X (gfx1250) — compile-verified, matrix operand order checked by reading, not run on hardware
//
#include <hip/hip_runtime.h>

typedef __attribute__((ext_vector_type(16))) __bf16 v16bf;
typedef __attribute__((ext_vector_type(8)))  float  v8f;
typedef __attribute__((ext_vector_type(4)))  unsigned int v4u;
typedef __attribute__((ext_vector_type(8)))  int v8i;
typedef __attribute__((ext_vector_type(4)))  int v4i;

union Frag32 { v16bf bf; float4 f4[2]; };

constexpr int    BATCH   = 16384;
constexpr int    NJOINT  = 12;
constexpr int    DCH     = 128;
constexpr size_t PRE_ELE = (size_t)BATCH * NJOINT * DCH;   // 25165824 elems

// workspace layout (bytes)
constexpr size_t OFF_STATS1 = 0;        // 512 f32
constexpr size_t OFF_STATS2 = 2048;     // 1024 f32
constexpr size_t OFF_SB1    = 6144;     // 512 f32
constexpr size_t OFF_SB2    = 8192;     // 1024 f32
constexpr size_t OFF_WBF    = 12288;    // 245760 bf16
constexpr size_t OFF_EFPRE  = 524288;   // bf16 PRE_ELE
// vfPre, efPost, vfPost follow; stage2 fp32 overlaps efPre/vfPre (dead by then)

// weight element offsets inside wbf
constexpr int WEV_O = 0, WVE_O = 32768, W1_O = 65536, W2_O = 114688,
              W3_O = 163840, W4_O = 196608, WTOT = 245760;

// finalize mapping: out joint a <- res[inv(a)]
__device__ const int t_kind[12] = {0,0,0,0,1,1,0,0,0,0,1,1};   // 1 = vf passthrough
__device__ const int t_src[12]  = {0,1,3,4,2,4,2,7,5,6,9,11};  // group idx or vf joint
__device__ const int t_bn[12]   = {0,1,2,2,0,0,1,3,2,2,0,0};

__global__ void k_zero(float* p, int n) {
    int i = blockIdx.x * blockDim.x + threadIdx.x;
    if (i < n) p[i] = 0.f;
}

__global__ void k_wcvt(const float* __restrict__ wev, const float* __restrict__ wve,
                       const float* __restrict__ w1,  const float* __restrict__ w2,
                       const float* __restrict__ w3,  const float* __restrict__ w4,
                       __bf16* __restrict__ dst) {
    int i = blockIdx.x * blockDim.x + threadIdx.x;
    if (i >= WTOT) return;
    float v;
    if      (i < WVE_O) v = wev[i];
    else if (i < W1_O)  v = wve[i - WVE_O];
    else if (i < W2_O)  v = w1[i - W1_O];
    else if (i < W3_O)  v = w2[i - W2_O];
    else if (i < W4_O)  v = w3[i - W3_O];
    else                v = w4[i - W4_O];
    dst[i] = (__bf16)v;
}

// ---------------- Stage 1: joint mix + dual GEMM (ef, vf) + BN stats ----------------
__global__ __launch_bounds__(256) void k_stage1(
    const float* __restrict__ gout, const float* __restrict__ eout,
    const float* __restrict__ evs,  const float* __restrict__ ves,
    const __bf16* __restrict__ wbf,
    __bf16* __restrict__ efPre, __bf16* __restrict__ vfPre,
    float* __restrict__ gStats1)
{
    constexpr int XS = 264;  // 256 + pad, keeps 16B alignment (528B rows)
    __shared__ __bf16 sGout[96 * 128];
    __shared__ alignas(16) __bf16 sXef[96 * XS];   // [eout | ev_g]
    __shared__ alignas(16) __bf16 sXve[96 * XS];   // [ve_g | eout]
    __shared__ float sShift[288];
    __shared__ float sSum[512];

    const int tid = threadIdx.x;
    const int b0  = blockIdx.x * 8;                // 8 batch elems -> 96 rows

    for (int i = tid; i < 512; i += 256) sSum[i] = 0.f;
    if (tid < 144) { sShift[tid] = evs[tid]; sShift[144 + tid] = ves[tid]; }

    for (int i = tid; i < 96 * 128; i += 256) {
        int r = i >> 7, d = i & 127;
        size_t gi = (size_t)(b0 * 12 + r) * 128 + d;
        sGout[r * 128 + d] = (__bf16)gout[gi];
        float ev = eout[gi];
        sXef[r * XS + d]       = (__bf16)ev;
        sXve[r * XS + 128 + d] = (__bf16)ev;
    }
    __syncthreads();

    for (int i = tid; i < 96 * 128; i += 256) {
        int r = i >> 7, d = i & 127;
        int bb = (r / 12) * 12, n = r - bb;
        float aE = 0.f, aV = 0.f;
#pragma unroll
        for (int j = 0; j < 12; j++) {
            float gv = (float)sGout[(bb + j) * 128 + d];
            aE += sShift[n * 12 + j] * gv;
            aV += sShift[144 + n * 12 + j] * gv;
        }
        sXef[r * XS + 128 + d] = (__bf16)aE;
        sXve[r * XS + d]       = (__bf16)aV;
    }
    __syncthreads();

    const int wave = tid >> 5, lane = tid & 31;
    const int lidx  = lane & 15;
    const int halfk = (lane & 16) ? 8 : 0;
    const int rtop  = (lane & 16) ? 8 : 0;
    const int nt    = wave;                 // each wave owns one N-tile

    for (int half = 0; half < 2; half++) {  // 0 = ef, 1 = vf
        // hoist all B fragments for this weight matrix into registers
        const __bf16* W = wbf + (half ? WVE_O : WEV_O);
        Frag32 bfr[8];
#pragma unroll
        for (int kk = 0; kk < 8; kk++) {
            const __bf16* bp = W + (nt * 16 + lidx) * 256 + kk * 32 + halfk;
            bfr[kk].f4[0] = *(const float4*)bp;
            bfr[kk].f4[1] = *(const float4*)(bp + 16);
        }
        const __bf16* X   = half ? sXve : sXef;
        __bf16*       dst = half ? vfPre : efPre;
        for (int mt = 0; mt < 6; mt++) {
            const __bf16* ap0 = X + (mt * 16 + lidx) * XS + halfk;
            v8f acc = {0.f,0.f,0.f,0.f,0.f,0.f,0.f,0.f};
#pragma unroll
            for (int kk = 0; kk < 8; kk++) {
                Frag32 a;
                a.f4[0] = *(const float4*)(ap0 + kk * 32);
                a.f4[1] = *(const float4*)(ap0 + kk * 32 + 16);
                acc = __builtin_amdgcn_wmma_f32_16x16x32_bf16(
                    false, a.bf, false, bfr[kk].bf, (short)0, acc, false, false);
            }
            int colc = nt * 16 + lidx;
            size_t row0 = (size_t)b0 * 12 + mt * 16 + rtop;
            float s = 0.f, q = 0.f;
#pragma unroll
            for (int r = 0; r < 8; r++) {
                float v = acc[r];
                s += v; q += v * v;
                dst[(row0 + r) * 128 + colc] = (__bf16)v;
            }
            atomicAdd(&sSum[half * 256 + colc], s);
            atomicAdd(&sSum[half * 256 + 128 + colc], q);
        }
    }
    __syncthreads();
    for (int c = tid; c < 512; c += 256) atomicAdd(&gStats1[c], sSum[c]);
}

__global__ void k_bnparams1(const float* __restrict__ gStats1,
                            const float* gev, const float* bev,
                            const float* gve, const float* bve,
                            float* __restrict__ sb1) {
    int t = threadIdx.x;            // 256
    int which = t >> 7, c = t & 127;
    float N = (float)((size_t)BATCH * 12);
    float s = gStats1[which * 256 + c], q = gStats1[which * 256 + 128 + c];
    float mean = s / N;
    float var  = q / N - mean * mean;
    float g = which ? gve[c] : gev[c];
    float b = which ? bve[c] : bev[c];
    float scale = g * rsqrtf(var + 1e-5f);
    sb1[which * 256 + c]       = scale;
    sb1[which * 256 + 128 + c] = b - mean * scale;
}

// BN+ReLU, re-layout to joint-major [12][B][128] bf16
__global__ __launch_bounds__(256) void k_bnrelu1(
    const __bf16* __restrict__ efPre, const __bf16* __restrict__ vfPre,
    const float* __restrict__ sb1,
    __bf16* __restrict__ efPost, __bf16* __restrict__ vfPost) {
    int chunk = blockIdx.x * 256 + threadIdx.x;
    const int perT = (int)(PRE_ELE / 8);
    int which = chunk >= perT;
    chunk -= which * perT;
    const __bf16* pre  = which ? vfPre  : efPre;
    __bf16*       post = which ? vfPost : efPost;
    size_t base = (size_t)chunk * 8;
    int b   = (int)(base / 1536);
    int rem = (int)(base - (size_t)b * 1536);
    int n = rem >> 7, d = rem & 127;
    float4 raw = *(const float4*)(pre + base);
    const __bf16* rb = (const __bf16*)&raw;
    alignas(16) __bf16 outv[8];
#pragma unroll
    for (int k = 0; k < 8; k++) {
        int c = d + k;
        float v = (float)rb[k];
        v = v * sb1[which * 256 + c] + sb1[which * 256 + 128 + c];
        outv[k] = (__bf16)fmaxf(v, 0.f);
    }
    *(float4*)(post + ((size_t)n * BATCH + b) * 128 + d) = *(const float4*)outv;
}

// ---------------- Stage 2: 8 gathered GEMM groups + BN stats (TDM-staged A) ----------------
__global__ __launch_bounds__(256) void k_stage2(
    const __bf16* __restrict__ efPost, const __bf16* __restrict__ vfPost,
    const __bf16* __restrict__ wbf,
    float* __restrict__ pre2, float* __restrict__ gStats2) {
    constexpr int RLEN = 136;            // 128 data elems + 8 pad elems (16B) per row
    constexpr int SLAB = 64 * RLEN;      // elems per 64-row slab block
    __shared__ alignas(16) __bf16 sA[3 * SLAB];
    __shared__ float sSum[256];
    const int tid = threadIdx.x;
    const int g   = blockIdx.y;
    const int b0  = blockIdx.x * 64;

    const int nslab = (g >= 3 && g <= 6) ? 2 : 3;
    const int K = nslab * 128;
    int st0, st1, st2 = 0, sj0, sj1, sj2 = 0;
    switch (g) {
        case 0:  st0=0; st1=0; st2=0; sj0=0;  sj1=1;  sj2=5;  break;
        case 1:  st0=1; st1=0; st2=0; sj0=0;  sj1=3;  sj2=7;  break;
        case 2:  st0=1; st1=0; st2=0; sj0=5;  sj1=7;  sj2=8;  break;
        case 3:  st0=1; st1=0;        sj0=1;  sj1=2;          break;
        case 4:  st0=1; st1=0;        sj0=3;  sj1=4;          break;
        case 5:  st0=1; st1=0;        sj0=8;  sj1=9;          break;
        case 6:  st0=1; st1=0;        sj0=10; sj1=11;         break;
        default: st0=1; st1=1; st2=0; sj0=6;  sj1=7;  sj2=10; break;
    }
    const int woff = (g == 0) ? W1_O : (g <= 2) ? W2_O : (g <= 6) ? W3_O : W4_O;
    const int bn   = (g == 0) ? 0 : (g <= 2) ? 1 : (g <= 6) ? 2 : 3;

    if (tid < 256) sSum[tid] = 0.f;

    // wave 0 issues one Tensor-DMA per slab: 64 rows x 256B global -> LDS with
    // 16B pad every 256B so rows land at a 272B (bank-friendly) stride.
    if (tid < 32) {
        for (int s = 0; s < nslab; s++) {
            int st = (s == 0) ? st0 : (s == 1) ? st1 : st2;
            int sj = (s == 0) ? sj0 : (s == 1) ? sj1 : sj2;
            const __bf16* src = (st ? vfPost : efPost) + ((size_t)sj * BATCH + b0) * 128;
            unsigned long long ga = (unsigned long long)(uintptr_t)src;
            unsigned ldsb = (unsigned)(uintptr_t)(sA + s * SLAB);
            v4u g0;
            g0[0] = 1u;                                   // count=1, load descriptor
            g0[1] = ldsb;                                 // lds_addr (bytes)
            g0[2] = (unsigned)(ga & 0xFFFFFFFFull);       // global_addr lo
            g0[3] = (unsigned)((ga >> 32) & 0x1FFFFFFull) | (2u << 30);  // hi | type=2
            v8i g1;
            g1[0] = (int)((3u << 16) | (1u << 20) | (5u << 22) | (3u << 25));
            //            data=8B     pad_en       ivl=64DW      amt=4DW
            g1[1] = (int)(32u << 16);   // tensor_dim0 = 32 (8B units) lo16 in [63:48]
            g1[2] = (int)(64u << 16);   // tensor_dim1 = 64 in [111:80] lo16
            g1[3] = (int)(32u << 16);   // tile_dim0 = 32 in [127:112]
            g1[4] = 64;                 // tile_dim1 = 64, tile_dim2 = 0
            g1[5] = 32;                 // tensor_dim0_stride = 32 (8B units)
            g1[6] = 0;
            g1[7] = 0;
            v4i z4 = {0, 0, 0, 0};
#if __clang_major__ >= 23
            v8i z8 = {0, 0, 0, 0, 0, 0, 0, 0};
            __builtin_amdgcn_tensor_load_to_lds(g0, g1, z4, z4, z8, 0);
#else
            __builtin_amdgcn_tensor_load_to_lds(g0, g1, z4, z4, 0);
#endif
        }
        __builtin_amdgcn_s_wait_tensorcnt(0);
    }

    const int wave = tid >> 5, lane = tid & 31;
    const int lidx  = lane & 15;
    const int halfk = (lane & 16) ? 8 : 0;
    const int rtop  = (lane & 16) ? 8 : 0;
    const int nt    = wave;
    const int ksteps = K >> 5;

    // hoist all B fragments for this wave's N-tile into registers
    const __bf16* W = wbf + woff;
    Frag32 bfr[12];
#pragma unroll
    for (int kk = 0; kk < 12; kk++) {
        if (kk < ksteps) {
            const __bf16* bp = W + (nt * 16 + lidx) * K + kk * 32 + halfk;
            bfr[kk].f4[0] = *(const float4*)bp;
            bfr[kk].f4[1] = *(const float4*)(bp + 16);
        }
    }
    __syncthreads();   // TDM data visible (wave0 waited tensorcnt before arriving)

    float* dst = pre2 + (size_t)g * BATCH * 128;
    for (int mt = 0; mt < 4; mt++) {
        v8f acc = {0.f,0.f,0.f,0.f,0.f,0.f,0.f,0.f};
#pragma unroll
        for (int kk = 0; kk < 12; kk++) {
            if (kk < ksteps) {
                int kb = kk * 32;
                int slab = kb >> 7;
                const __bf16* ap = sA + slab * SLAB + (mt * 16 + lidx) * RLEN
                                 + (kb & 127) + halfk;
                Frag32 a;
                a.f4[0] = *(const float4*)ap;
                a.f4[1] = *(const float4*)(ap + 16);
                acc = __builtin_amdgcn_wmma_f32_16x16x32_bf16(
                    false, a.bf, false, bfr[kk].bf, (short)0, acc, false, false);
            }
        }
        int colc = nt * 16 + lidx;
        size_t row0 = (size_t)b0 + mt * 16 + rtop;
        float s = 0.f, q = 0.f;
#pragma unroll
        for (int r = 0; r < 8; r++) {
            float v = acc[r];
            s += v; q += v * v;
            dst[(row0 + r) * 128 + colc] = v;
        }
        atomicAdd(&sSum[colc], s);
        atomicAdd(&sSum[128 + colc], q);
    }
    __syncthreads();
    if (tid < 256) atomicAdd(&gStats2[bn * 256 + tid], sSum[tid]);
}

__global__ void k_bnparams2(const float* __restrict__ gStats2,
                            const float* g1, const float* b1, const float* g2, const float* b2,
                            const float* g3, const float* b3, const float* g4, const float* b4,
                            float* __restrict__ sb2) {
    int t = threadIdx.x;             // 512
    int set = t >> 7, c = t & 127;
    float cnt = (set == 1) ? 2.f * BATCH : (set == 2) ? 4.f * BATCH : (float)BATCH;
    float s = gStats2[set * 256 + c], q = gStats2[set * 256 + 128 + c];
    float mean = s / cnt, var = q / cnt - mean * mean;
    const float* gg = (set == 0) ? g1 : (set == 1) ? g2 : (set == 2) ? g3 : g4;
    const float* bb = (set == 0) ? b1 : (set == 1) ? b2 : (set == 2) ? b3 : b4;
    float scale = gg[c] * rsqrtf(var + 1e-5f);
    sb2[set * 256 + c]       = scale;
    sb2[set * 256 + 128 + c] = bb[c] - mean * scale;
}

__global__ __launch_bounds__(256) void k_final(
    const float* __restrict__ gout, const float* __restrict__ pre2,
    const __bf16* __restrict__ vfPost, const float* __restrict__ sb2,
    float* __restrict__ out) {
    int chunk = blockIdx.x * 256 + threadIdx.x;   // < PRE_ELE/8
    size_t base = (size_t)chunk * 8;
    int b   = (int)(base / 1536);
    int rem = (int)(base - (size_t)b * 1536);
    int a = rem >> 7, d = rem & 127;
    int kind = t_kind[a], src = t_src[a], bnv = t_bn[a];
    float res[8];
    if (!kind) {
        const float* p = pre2 + ((size_t)src * BATCH + b) * 128 + d;
#pragma unroll
        for (int k = 0; k < 8; k++)
            res[k] = p[k] * sb2[bnv * 256 + d + k] + sb2[bnv * 256 + 128 + d + k];
    } else {
        const __bf16* p = vfPost + ((size_t)src * BATCH + b) * 128 + d;
        float4 raw = *(const float4*)p;
        const __bf16* rb = (const __bf16*)&raw;
#pragma unroll
        for (int k = 0; k < 8; k++) res[k] = (float)rb[k];
    }
#pragma unroll
    for (int k = 0; k < 8; k++)
        out[base + k] = fmaxf(gout[base + k] + res[k], 0.f);
}

extern "C" void kernel_launch(void* const* d_in, const int* in_sizes, int n_in,
                              void* d_out, int out_size, void* d_ws, size_t ws_size,
                              hipStream_t stream) {
    (void)in_sizes; (void)n_in; (void)out_size; (void)ws_size;
    const float* gout = (const float*)d_in[0];
    const float* eout = (const float*)d_in[1];
    const float* evs  = (const float*)d_in[2];
    const float* ves  = (const float*)d_in[3];
    const float* Wev  = (const float*)d_in[4];
    const float* Wve  = (const float*)d_in[5];
    const float* W1   = (const float*)d_in[6];
    const float* W2   = (const float*)d_in[7];
    const float* W3   = (const float*)d_in[8];
    const float* W4   = (const float*)d_in[9];
    const float* gev  = (const float*)d_in[10];
    const float* bev  = (const float*)d_in[11];
    const float* gve  = (const float*)d_in[12];
    const float* bve  = (const float*)d_in[13];
    const float* g1   = (const float*)d_in[14];
    const float* b1   = (const float*)d_in[15];
    const float* g2   = (const float*)d_in[16];
    const float* b2   = (const float*)d_in[17];
    const float* g3   = (const float*)d_in[18];
    const float* b3   = (const float*)d_in[19];
    const float* g4   = (const float*)d_in[20];
    const float* b4   = (const float*)d_in[21];

    char* ws = (char*)d_ws;
    float*  gStats1 = (float*)(ws + OFF_STATS1);
    float*  gStats2 = (float*)(ws + OFF_STATS2);
    float*  sb1     = (float*)(ws + OFF_SB1);
    float*  sb2     = (float*)(ws + OFF_SB2);
    __bf16* wbf     = (__bf16*)(ws + OFF_WBF);
    __bf16* efPre   = (__bf16*)(ws + OFF_EFPRE);
    __bf16* vfPre   = efPre + PRE_ELE;
    __bf16* efPost  = vfPre + PRE_ELE;
    __bf16* vfPost  = efPost + PRE_ELE;
    float*  pre2    = (float*)efPre;   // overlap: efPre/vfPre dead after k_bnrelu1

    k_zero<<<12, 256, 0, stream>>>((float*)ws, 3072);
    k_wcvt<<<(WTOT + 255) / 256, 256, 0, stream>>>(Wev, Wve, W1, W2, W3, W4, wbf);
    k_stage1<<<BATCH / 8, 256, 0, stream>>>(gout, eout, evs, ves, wbf, efPre, vfPre, gStats1);
    k_bnparams1<<<1, 256, 0, stream>>>(gStats1, gev, bev, gve, bve, sb1);
    k_bnrelu1<<<(int)(2 * PRE_ELE / 8 / 256), 256, 0, stream>>>(efPre, vfPre, sb1, efPost, vfPost);
    k_stage2<<<dim3(BATCH / 64, 8), 256, 0, stream>>>(efPost, vfPost, wbf, pre2, gStats2);
    k_bnparams2<<<1, 512, 0, stream>>>(gStats2, g1, b1, g2, b2, g3, b3, g4, b4, sb2);
    k_final<<<(int)(PRE_ELE / 8 / 256), 256, 0, stream>>>(gout, pre2, vfPost, sb2, (float*)d_out);
}
